// CausalSelfAttention_80479097192777
// MI455X (gfx1250) — matrix operand order repaired and checked by reading, not compiled, not run on hardware
//
#include <hip/hip_runtime.h>

typedef __bf16 bf16_t;
typedef __attribute__((ext_vector_type(16))) __bf16 v16bf;
typedef __attribute__((ext_vector_type(8)))  __bf16 v8bf;
typedef __attribute__((ext_vector_type(8)))  float  v8f;

union BF16x16 { v16bf v; v8bf h[2]; };

#define B_  4
#define T_  2048
#define C_  1024
#define H_  16
#define HD_ 64

__device__ __forceinline__ v8f wmma_bf16(v16bf a, v16bf b, v8f c) {
  return __builtin_amdgcn_wmma_f32_16x16x32_bf16(false, a, false, b, (short)0, c,
                                                 false, false);
}

// Async copy 16 bytes global -> LDS (ASYNCcnt-tracked, CDNA5 VFLAT async op).
__device__ __forceinline__ void async_ld16(unsigned lds_off, const bf16_t* g) {
  asm volatile("global_load_async_to_lds_b128 %0, %1, off"
               :: "v"(lds_off), "v"((unsigned long long)(uintptr_t)g)
               : "memory");
}

// ---------------- prep kernels ----------------
__global__ void __launch_bounds__(256) k_cvt(const float* __restrict__ in,
                                             bf16_t* __restrict__ out, int n) {
  int i = blockIdx.x * 256 + threadIdx.x;
  if (i < n) out[i] = (bf16_t)in[i];
}

// out[n*K + k] = in[k*N + n]  (produce B^T so WMMA B-frags are K-contiguous)
__global__ void __launch_bounds__(256) k_transpose_cvt(const float* __restrict__ in,
                                                       bf16_t* __restrict__ out,
                                                       int K, int N) {
  int i = blockIdx.x * 256 + threadIdx.x;
  if (i < K * N) {
    int n = i / K;
    int k = i - n * K;
    out[i] = (bf16_t)in[k * N + n];
  }
}

// ---------------- QKV GEMM:  qkv = x @ w_attn + b_attn, scatter to q/k/vT ----------------
__global__ void __launch_bounds__(256) k_qkv_gemm(const bf16_t* __restrict__ xb,
                                                  const bf16_t* __restrict__ wta,
                                                  const float*  __restrict__ bias,
                                                  bf16_t* __restrict__ qb,
                                                  bf16_t* __restrict__ kbuf,
                                                  bf16_t* __restrict__ vt) {
  const int lane = threadIdx.x & 31;
  const int hi = lane >> 4, ln = lane & 15;
  const int wid = threadIdx.x >> 5;
  const int wm = wid & 1, wn = wid >> 1;
  const int m0 = blockIdx.x * 128 + wm * 64;   // rows of x (B*T)
  const int n0 = blockIdx.y * 256 + wn * 64;   // cols of qkv (3C)

  v8f acc[4][4] = {};
  for (int k0 = 0; k0 < C_; k0 += 32) {
    v16bf a[4], bm[4];
#pragma unroll
    for (int mt = 0; mt < 4; ++mt) {
      const bf16_t* p = xb + (size_t)(m0 + mt * 16 + ln) * C_ + k0;
      BF16x16 u;
      u.h[0] = *(const v8bf*)(p + hi * 8);
      u.h[1] = *(const v8bf*)(p + 16 + hi * 8);
      a[mt] = u.v;
    }
#pragma unroll
    for (int nt = 0; nt < 4; ++nt) {
      const bf16_t* p = wta + (size_t)(n0 + nt * 16 + ln) * C_ + k0;
      BF16x16 u;
      u.h[0] = *(const v8bf*)(p + hi * 8);
      u.h[1] = *(const v8bf*)(p + 16 + hi * 8);
      bm[nt] = u.v;
    }
#pragma unroll
    for (int mt = 0; mt < 4; ++mt)
#pragma unroll
      for (int nt = 0; nt < 4; ++nt)
        acc[mt][nt] = wmma_bf16(a[mt], bm[nt], acc[mt][nt]);
  }

#pragma unroll
  for (int mt = 0; mt < 4; ++mt)
#pragma unroll
    for (int nt = 0; nt < 4; ++nt) {
      const int Cc = n0 + nt * 16 + ln;
      const float bv = bias[Cc];
      const int which = Cc >> 10, cc = Cc & 1023;
      const int h = cc >> 6, d = cc & 63;
#pragma unroll
      for (int e = 0; e < 8; ++e) {
        const int R = m0 + mt * 16 + e + hi * 8;
        const int b = R >> 11, t = R & 2047;
        const float v = acc[mt][nt][e] + bv;
        const size_t bh = (size_t)(b * H_ + h);
        if (which == 0)      qb  [(bh * T_ + t) * HD_ + d] = (bf16_t)(v * 0.125f);
        else if (which == 1) kbuf[(bh * T_ + t) * HD_ + d] = (bf16_t)v;
        else                 vt  [(bh * HD_ + d) * T_ + t] = (bf16_t)v;  // V^T
      }
    }
}

// ---------------- flash attention:  y = softmax(mask(Q K^T)) V ----------------
// K/V tiles double-buffered in LDS via async global->LDS copies.
__global__ void __launch_bounds__(256) k_flash_attn(const bf16_t* __restrict__ qb,
                                                    const bf16_t* __restrict__ kbuf,
                                                    const bf16_t* __restrict__ vt,
                                                    bf16_t* __restrict__ yb) {
  __shared__ __attribute__((aligned(16))) bf16_t Kt[2][32][64];   // keys x d
  __shared__ __attribute__((aligned(16))) bf16_t Vt2[2][64][32];  // d x keys (V^T)
  __shared__ __attribute__((aligned(16))) bf16_t Plds[8][16][32]; // per-wave P tile

  const int tid = threadIdx.x;
  const int lane = tid & 31;
  const int hi = lane >> 4, ln = lane & 15;
  const int wid = tid >> 5;
  const int bh = blockIdx.y;
  const int b = bh >> 4, h = bh & 15;
  const int q0 = blockIdx.x * 128;
  const int qrow = q0 + wid * 16;

  const bf16_t* Qp = qb   + (size_t)bh * T_ * HD_;
  const bf16_t* Kp = kbuf + (size_t)bh * T_ * HD_;
  const bf16_t* Vp = vt   + (size_t)bh * HD_ * T_;

  // per-thread staging chunk: 16B each into K tile and V tile
  const int krow = tid >> 3, kcol = (tid & 7) * 8;   // 32 rows x 8 chunks
  const int vrow = tid >> 2, vcol = (tid & 3) * 8;   // 64 rows x 4 chunks
  unsigned ldsK[2], ldsV[2];
  ldsK[0] = (unsigned)(uintptr_t)&Kt[0][krow][kcol];
  ldsK[1] = (unsigned)(uintptr_t)&Kt[1][krow][kcol];
  ldsV[0] = (unsigned)(uintptr_t)&Vt2[0][vrow][vcol];
  ldsV[1] = (unsigned)(uintptr_t)&Vt2[1][vrow][vcol];

  // Q A-fragments (persist across the key loop); softmax scale folded into Q
  v16bf aQ[2];
#pragma unroll
  for (int kk = 0; kk < 2; ++kk) {
    const bf16_t* p = Qp + (size_t)(qrow + ln) * HD_ + 32 * kk;
    BF16x16 u;
    u.h[0] = *(const v8bf*)(p + hi * 8);
    u.h[1] = *(const v8bf*)(p + 16 + hi * 8);
    aQ[kk] = u.v;
  }

  v8f o[4] = {};
  float mrow[8], lrow[8];
#pragma unroll
  for (int e = 0; e < 8; ++e) { mrow[e] = -1e30f; lrow[e] = 0.0f; }

  const int nblk = (q0 + 128) / 32;  // causal: keys < q0+128

  // prologue: stage first K/V tile
  async_ld16(ldsK[0], Kp + (size_t)krow * HD_ + kcol);
  async_ld16(ldsV[0], Vp + (size_t)vrow * T_ + vcol);

  for (int i = 0; i < nblk; ++i) {
    const int kb0 = i * 32;
    if (i + 1 < nblk) {  // prefetch next tile into other buffer
      const int kn = kb0 + 32;
      async_ld16(ldsK[(i + 1) & 1], Kp + (size_t)(kn + krow) * HD_ + kcol);
      async_ld16(ldsV[(i + 1) & 1], Vp + (size_t)vrow * T_ + kn + vcol);
      // async loads complete in order: <=2 outstanding => current tile landed
      asm volatile("s_wait_asynccnt 0x2" ::: "memory");
    } else {
      asm volatile("s_wait_asynccnt 0x0" ::: "memory");
    }
    __syncthreads();

    const bf16_t (*K_)[64] = Kt[i & 1];
    const bf16_t (*V_)[32] = Vt2[i & 1];

    // S = Q K^T for two 16-wide key chunks (B-frags from LDS)
    v8f s[2] = {};
#pragma unroll
    for (int nc = 0; nc < 2; ++nc) {
      const bf16_t* kp = K_[nc * 16 + ln];
#pragma unroll
      for (int kk = 0; kk < 2; ++kk) {
        BF16x16 u;
        u.h[0] = *(const v8bf*)(kp + 32 * kk + hi * 8);
        u.h[1] = *(const v8bf*)(kp + 32 * kk + 16 + hi * 8);
        s[nc] = wmma_bf16(aQ[kk], u.v, s[nc]);
      }
    }

    // causal mask + online softmax (C layout: each 16-lane group shares 8 rows)
    const int key0 = kb0 + ln, key1 = kb0 + 16 + ln;
#pragma unroll
    for (int e = 0; e < 8; ++e) {
      const int row = qrow + e + hi * 8;
      float s0 = (key0 <= row) ? s[0][e] : -1e30f;
      float s1 = (key1 <= row) ? s[1][e] : -1e30f;
      float rm = fmaxf(s0, s1);
#pragma unroll
      for (int off = 1; off < 16; off <<= 1)
        rm = fmaxf(rm, __shfl_xor(rm, off, 16));
      const float mn = fmaxf(mrow[e], rm);
      const float alpha = __expf(mrow[e] - mn);
      const float p0 = __expf(s0 - mn);
      const float p1 = __expf(s1 - mn);
      float ps = p0 + p1;
#pragma unroll
      for (int off = 1; off < 16; off <<= 1)
        ps += __shfl_xor(ps, off, 16);
      lrow[e] = lrow[e] * alpha + ps;
      mrow[e] = mn;
#pragma unroll
      for (int nd = 0; nd < 4; ++nd) o[nd][e] *= alpha;
      Plds[wid][e + hi * 8][ln]      = (bf16_t)p0;
      Plds[wid][e + hi * 8][16 + ln] = (bf16_t)p1;
    }
    // reload P as an A-fragment (wave-private LDS region, wave-synchronous)
    BF16x16 up;
    up.h[0] = *(const v8bf*)&Plds[wid][ln][hi * 8];
    up.h[1] = *(const v8bf*)&Plds[wid][ln][16 + hi * 8];

    // O += P @ V  (B-frags from LDS V^T tile)
#pragma unroll
    for (int nd = 0; nd < 4; ++nd) {
      const bf16_t* vp = V_[nd * 16 + ln];
      BF16x16 uv;
      uv.h[0] = *(const v8bf*)(vp + hi * 8);
      uv.h[1] = *(const v8bf*)(vp + 16 + hi * 8);
      o[nd] = wmma_bf16(up.v, uv.v, o[nd]);
    }
    __syncthreads();  // all reads of this tile done before it is overwritten
  }

  // normalize and store y in [B,T,C] bf16
#pragma unroll
  for (int e = 0; e < 8; ++e) {
    const float inv = 1.0f / lrow[e];
    const int t = qrow + e + hi * 8;
#pragma unroll
    for (int nd = 0; nd < 4; ++nd) {
      const int c = h * HD_ + nd * 16 + ln;
      yb[((size_t)b * T_ + t) * C_ + c] = (bf16_t)(o[nd][e] * inv);
    }
  }
}

// ---------------- output projection:  out = y @ w_proj + b_proj (fp32 out) ----------------
__global__ void __launch_bounds__(256) k_proj_gemm(const bf16_t* __restrict__ yb,
                                                   const bf16_t* __restrict__ wtp,
                                                   const float*  __restrict__ bias,
                                                   float* __restrict__ out) {
  const int lane = threadIdx.x & 31;
  const int hi = lane >> 4, ln = lane & 15;
  const int wid = threadIdx.x >> 5;
  const int wm = wid & 1, wn = wid >> 1;
  const int m0 = blockIdx.x * 128 + wm * 64;
  const int n0 = blockIdx.y * 256 + wn * 64;

  v8f acc[4][4] = {};
  for (int k0 = 0; k0 < C_; k0 += 32) {
    v16bf a[4], bm[4];
#pragma unroll
    for (int mt = 0; mt < 4; ++mt) {
      const bf16_t* p = yb + (size_t)(m0 + mt * 16 + ln) * C_ + k0;
      BF16x16 u;
      u.h[0] = *(const v8bf*)(p + hi * 8);
      u.h[1] = *(const v8bf*)(p + 16 + hi * 8);
      a[mt] = u.v;
    }
#pragma unroll
    for (int nt = 0; nt < 4; ++nt) {
      const bf16_t* p = wtp + (size_t)(n0 + nt * 16 + ln) * C_ + k0;
      BF16x16 u;
      u.h[0] = *(const v8bf*)(p + hi * 8);
      u.h[1] = *(const v8bf*)(p + 16 + hi * 8);
      bm[nt] = u.v;
    }
#pragma unroll
    for (int mt = 0; mt < 4; ++mt)
#pragma unroll
      for (int nt = 0; nt < 4; ++nt)
        acc[mt][nt] = wmma_bf16(a[mt], bm[nt], acc[mt][nt]);
  }

#pragma unroll
  for (int mt = 0; mt < 4; ++mt)
#pragma unroll
    for (int nt = 0; nt < 4; ++nt) {
      const int Cc = n0 + nt * 16 + ln;
      const float bv = bias[Cc];
#pragma unroll
      for (int e = 0; e < 8; ++e) {
        const int R = m0 + mt * 16 + e + hi * 8;
        out[(size_t)R * C_ + Cc] = acc[mt][nt][e] + bv;
      }
    }
}

// ---------------- host launcher ----------------
extern "C" void kernel_launch(void* const* d_in, const int* in_sizes, int n_in,
                              void* d_out, int out_size, void* d_ws, size_t ws_size,
                              hipStream_t stream) {
  const float* x      = (const float*)d_in[0];
  const float* w_attn = (const float*)d_in[1];
  const float* b_attn = (const float*)d_in[2];
  const float* w_proj = (const float*)d_in[3];
  const float* b_proj = (const float*)d_in[4];
  float* out = (float*)d_out;

  char* ws = (char*)d_ws;
  size_t off = 0;
  auto take = [&](size_t elems) -> bf16_t* {
    off = (off + 255) & ~(size_t)255;
    bf16_t* p = (bf16_t*)(ws + off);
    off += elems * sizeof(bf16_t);
    return p;
  };
  const size_t nx = (size_t)B_ * T_ * C_;          // 8M elems
  bf16_t* xb  = take(nx);                          // x in bf16
  bf16_t* wta = take((size_t)3 * C_ * C_);         // w_attn^T bf16 [3C][C]
  bf16_t* wtp = take((size_t)C_ * C_);             // w_proj^T bf16 [C][C]
  bf16_t* qb  = take(nx);                          // Q [B,H,T,hd] (pre-scaled)
  bf16_t* kb  = take(nx);                          // K [B,H,T,hd]
  bf16_t* vt  = take(nx);                          // V^T [B,H,hd,T]
  bf16_t* yb  = take(nx);                          // attention out [B,T,C]

  k_cvt<<<dim3((unsigned)((nx + 255) / 256)), dim3(256), 0, stream>>>(x, xb, (int)nx);
  k_transpose_cvt<<<dim3((3 * C_ * C_ + 255) / 256), dim3(256), 0, stream>>>(w_attn, wta, C_, 3 * C_);
  k_transpose_cvt<<<dim3((C_ * C_ + 255) / 256), dim3(256), 0, stream>>>(w_proj, wtp, C_, C_);
  k_qkv_gemm<<<dim3(64, 12), dim3(256), 0, stream>>>(xb, wta, b_attn, qb, kb, vt);
  k_flash_attn<<<dim3(T_ / 128, B_ * H_), dim3(256), 0, stream>>>(qb, kb, vt, yb);
  k_proj_gemm<<<dim3(64, 4), dim3(256), 0, stream>>>(yb, wtp, b_proj, out);
}
